// CharacterCNNEmbedding_18511309046095
// MI455X (gfx1250) — matrix operand order repaired and checked by reading, not compiled, not run on hardware
//
#include <hip/hip_runtime.h>
#include <hip/hip_bf16.h>
#include <cstdint>
#include <cstddef>

typedef __attribute__((ext_vector_type(16))) __bf16 v16bf;
typedef __attribute__((ext_vector_type(8)))  float  v8f;

#define NTOK 4096
#define FDIM 2048
#define KC   128
#define CLEN 50

__device__ __forceinline__ unsigned short f2bf(float x) {
    union { float f; unsigned u; } c; c.f = x;
    unsigned u = c.u;
    u += 0x7FFFu + ((u >> 16) & 1u);          // round-to-nearest-even
    return (unsigned short)(u >> 16);
}

union Frag { v16bf v; uint4 u[2]; };

__device__ __forceinline__ v8f wmma_bf16(const Frag& a, const Frag& b, v8f c) {
    return __builtin_amdgcn_wmma_f32_16x16x32_bf16(
        false, a.v, false, b.v, (short)0, c, false, false);
}

__device__ __forceinline__ int filt_width(int f) {
    if (f < 32)   return 1;
    if (f < 64)   return 2;
    if (f < 128)  return 3;
    if (f < 256)  return 4;
    if (f < 512)  return 5;
    if (f < 1024) return 6;
    return 7;
}

// ---------------- weight prep ----------------

__global__ void cvt_bf16_kernel(const float* __restrict__ src,
                                unsigned short* __restrict__ dst, int n) {
    int i = blockIdx.x * blockDim.x + threadIdx.x;
    int stride = gridDim.x * blockDim.x;
    for (; i < n; i += stride) dst[i] = f2bf(src[i]);
}

// Wcat[f][tap*16+c] = conv_w(f)[c][tap] for tap < width(f), else 0.  bcat[f] = bias.
__global__ void build_wcat_kernel(
    const float* __restrict__ w0, const float* __restrict__ w1,
    const float* __restrict__ w2, const float* __restrict__ w3,
    const float* __restrict__ w4, const float* __restrict__ w5,
    const float* __restrict__ w6,
    const float* __restrict__ b0, const float* __restrict__ b1,
    const float* __restrict__ b2, const float* __restrict__ b3,
    const float* __restrict__ b4, const float* __restrict__ b5,
    const float* __restrict__ b6,
    unsigned short* __restrict__ wcat, float* __restrict__ bcat)
{
    int i = blockIdx.x * blockDim.x + threadIdx.x;
    if (i >= FDIM * KC) return;
    int f = i >> 7, kk = i & 127;
    int tap = kk >> 4, c = kk & 15;
    const float* w; const float* b; int fl, wid;
    if      (f < 32)   { w = w0; b = b0; fl = f;        wid = 1; }
    else if (f < 64)   { w = w1; b = b1; fl = f - 32;   wid = 2; }
    else if (f < 128)  { w = w2; b = b2; fl = f - 64;   wid = 3; }
    else if (f < 256)  { w = w3; b = b3; fl = f - 128;  wid = 4; }
    else if (f < 512)  { w = w4; b = b4; fl = f - 256;  wid = 5; }
    else if (f < 1024) { w = w5; b = b5; fl = f - 512;  wid = 6; }
    else               { w = w6; b = b6; fl = f - 1024; wid = 7; }
    float val = 0.f;
    if (tap < wid) val = w[(fl * 16 + c) * wid + tap];   // [nf][CDIM][w]
    wcat[i] = f2bf(val);
    if (kk == 0) bcat[f] = b[fl];
}

// ---------------- embedding + conv + relu + max-over-time ----------------
// One workgroup (8 waves) per token.  im2col A[64x128] bf16 in LDS, then
// A @ Wcat^T via v_wmma_f32_16x16x32_bf16; masked relu-max epilogue.
// B fragments for the next N-tile are prefetched before the WMMA chain.

__device__ __forceinline__ void load_bnt(const unsigned short* __restrict__ wcat,
                                         int fcol, int half, Frag b[4]) {
#pragma unroll
    for (int ks = 0; ks < 4; ++ks) {
        const uint4* bp = (const uint4*)(wcat + (size_t)fcol * KC + ks * 32 + half * 16);
        b[ks].u[0] = bp[0]; b[ks].u[1] = bp[1];
    }
}

__global__ __launch_bounds__(256)
void embed_conv_kernel(const int* __restrict__ char_ids,
                       const float* __restrict__ char_table,
                       const unsigned short* __restrict__ wcat,
                       const float* __restrict__ bcat,
                       float* __restrict__ hf,
                       unsigned short* __restrict__ hb)
{
    __shared__ unsigned short Aim[64 * KC];
    __shared__ int cid[64];
    const int tok = blockIdx.x;
    const int tid = threadIdx.x;

    if (tid < 64) cid[tid] = (tid < CLEN) ? char_ids[tok * CLEN + tid] : 0;
    __syncthreads();

    for (int i = tid; i < 64 * KC; i += 256) {
        int p = i >> 7, kk = i & 127, tap = kk >> 4, c = kk & 15;
        float v = 0.f;
        if (p < CLEN && tap < 7) {
            int ch = cid[p + tap];                // p+tap <= 55 < 64
            if (ch != 0) v = char_table[ch * 16 + c];   // padding_idx=0 -> zero row
        }
        Aim[i] = f2bf(v);
    }
    __syncthreads();

    const int lane = tid & 31;
    const int wave = tid >> 5;
    const int half = lane >> 4;
    const int lrow = lane & 15;

    Frag b[4];
    load_bnt(wcat, wave * 256 + lrow, half, b);    // prefetch nt = 0

    for (int nt = 0; nt < 16; ++nt) {
        const int fcol = wave * 256 + nt * 16 + lrow;   // D-matrix column = lane&15
        const int wlim = CLEN - filt_width(fcol);       // last valid position
        const float bias = bcat[fcol];

        Frag bn[4];
        if (nt < 15) load_bnt(wcat, fcol + 16, half, bn);   // prefetch next N-tile

        v8f acc[4] = {};
#pragma unroll
        for (int ks = 0; ks < 4; ++ks) {
#pragma unroll
            for (int m = 0; m < 4; ++m) {
                Frag a;
                const unsigned short* ap = Aim + (m * 16 + lrow) * KC + ks * 32 + half * 16;
                a.u[0] = *(const uint4*)ap;
                a.u[1] = *(const uint4*)(ap + 8);
                acc[m] = wmma_bf16(a, b[ks], acc[m]);
            }
        }
        float hmax = 0.f;
#pragma unroll
        for (int m = 0; m < 4; ++m)
#pragma unroll
            for (int v = 0; v < 8; ++v) {
                int p = m * 16 + v + half * 8;          // D row
                float val = (p <= wlim) ? fmaxf(acc[m][v] + bias, 0.f) : 0.f;
                hmax = fmaxf(hmax, val);
            }
        hmax = fmaxf(hmax, __shfl_xor(hmax, 16, 32));   // fold the two lane-halves
        if (half == 0) {
            hf[(size_t)tok * FDIM + fcol] = hmax;
            hb[(size_t)tok * FDIM + fcol] = f2bf(hmax);
        }
#pragma unroll
        for (int ks = 0; ks < 4; ++ks) b[ks] = bn[ks];
    }
}

// ---------------- fused highway layer (T and G GEMMs + gating) ----------------
// grid (8 Nblocks, 64 Mblocks), 256 threads.  Double-buffered 64x32 A panel in
// LDS (one barrier per K-step); next panel + next B fragments prefetched before
// the current 16 WMMAs so global loads overlap the matrix pipe.

__device__ __forceinline__ void load_b2(const unsigned short* __restrict__ w,
                                        int c0, int c1, int kcol, Frag b[2]) {
    const uint4* p0 = (const uint4*)(w + (size_t)c0 * FDIM + kcol);
    b[0].u[0] = p0[0]; b[0].u[1] = p0[1];
    const uint4* p1 = (const uint4*)(w + (size_t)c1 * FDIM + kcol);
    b[1].u[0] = p1[0]; b[1].u[1] = p1[1];
}

__global__ __launch_bounds__(256)
void highway_kernel(const unsigned short* __restrict__ hbin,
                    const float* __restrict__ hfin,
                    const unsigned short* __restrict__ tw, const float* __restrict__ tbv,
                    const unsigned short* __restrict__ gw, const float* __restrict__ gbv,
                    float* __restrict__ hfo, unsigned short* __restrict__ hbo)
{
    __shared__ unsigned short As[2][64 * 32];
    const int tid  = threadIdx.x;
    const int lane = tid & 31, wave = tid >> 5;
    const int half = lane >> 4, lrow = lane & 15;
    const int mbase = blockIdx.y * 64;
    const int nbase = blockIdx.x * 256 + wave * 32;
    const int srow = tid >> 2;              // staging: 64 rows x 32 bf16
    const int schunk = (tid & 3) * 8;
    const int c0 = nbase + lrow, c1 = nbase + 16 + lrow;
    const unsigned short* stgp = hbin + (size_t)(mbase + srow) * FDIM + schunk;

    v8f accT[4][2] = {};
    v8f accG[4][2] = {};

    // prologue: stage panel 0, load B frags for ks=0
    *(uint4*)(As[0] + srow * 32 + schunk) = *(const uint4*)stgp;
    Frag bT[2], bG[2];
    load_b2(tw, c0, c1, half * 16, bT);
    load_b2(gw, c0, c1, half * 16, bG);
    __syncthreads();

    for (int ks = 0; ks < 64; ++ks) {
        const int buf = ks & 1;
        Frag a[4];
#pragma unroll
        for (int m = 0; m < 4; ++m) {
            const unsigned short* ap = As[buf] + (m * 16 + lrow) * 32 + half * 16;
            a[m].u[0] = *(const uint4*)ap;
            a[m].u[1] = *(const uint4*)(ap + 8);
        }
        // prefetch next panel + next B fragments (overlap with WMMAs below)
        uint4 stgN;
        Frag nT[2], nG[2];
        if (ks < 63) {
            stgN = *(const uint4*)(stgp + (ks + 1) * 32);
            const int kcol = (ks + 1) * 32 + half * 16;
            load_b2(tw, c0, c1, kcol, nT);
            load_b2(gw, c0, c1, kcol, nG);
        }
#pragma unroll
        for (int m = 0; m < 4; ++m) {
            accT[m][0] = wmma_bf16(a[m], bT[0], accT[m][0]);
            accT[m][1] = wmma_bf16(a[m], bT[1], accT[m][1]);
            accG[m][0] = wmma_bf16(a[m], bG[0], accG[m][0]);
            accG[m][1] = wmma_bf16(a[m], bG[1], accG[m][1]);
        }
        if (ks < 63) {
            *(uint4*)(As[buf ^ 1] + srow * 32 + schunk) = stgN;
            __syncthreads();
            bT[0] = nT[0]; bT[1] = nT[1];
            bG[0] = nG[0]; bG[1] = nG[1];
        }
    }

    const float tb0 = tbv[c0], tb1 = tbv[c1];
    const float gb0 = gbv[c0], gb1 = gbv[c1];
#pragma unroll
    for (int m = 0; m < 4; ++m)
#pragma unroll
        for (int v = 0; v < 8; ++v) {
            const size_t r = (size_t)(mbase + m * 16 + v + half * 8);
            {
                float t = fmaxf(accT[m][0][v] + tb0, 0.f);
                float g = 1.f / (1.f + __expf(-(accG[m][0][v] + gb0)));
                float ho = hfin[r * FDIM + c0];
                float hn = g * t + (1.f - g) * ho;
                hfo[r * FDIM + c0] = hn;
                hbo[r * FDIM + c0] = f2bf(hn);
            }
            {
                float t = fmaxf(accT[m][1][v] + tb1, 0.f);
                float g = 1.f / (1.f + __expf(-(accG[m][1][v] + gb1)));
                float ho = hfin[r * FDIM + c1];
                float hn = g * t + (1.f - g) * ho;
                hfo[r * FDIM + c1] = hn;
                hbo[r * FDIM + c1] = f2bf(hn);
            }
        }
}

// ---------------- projection ----------------

__global__ __launch_bounds__(256)
void proj_kernel(const unsigned short* __restrict__ hbin,
                 const unsigned short* __restrict__ pw, const float* __restrict__ pb,
                 float* __restrict__ out)
{
    __shared__ unsigned short As[2][64 * 32];
    const int tid  = threadIdx.x;
    const int lane = tid & 31, wave = tid >> 5;
    const int half = lane >> 4, lrow = lane & 15;
    const int mbase = blockIdx.y * 64;
    const int nbase = blockIdx.x * 256 + wave * 32;
    const int srow = tid >> 2;
    const int schunk = (tid & 3) * 8;
    const int c0 = nbase + lrow, c1 = nbase + 16 + lrow;
    const unsigned short* stgp = hbin + (size_t)(mbase + srow) * FDIM + schunk;

    v8f acc[4][2] = {};

    *(uint4*)(As[0] + srow * 32 + schunk) = *(const uint4*)stgp;
    Frag b[2];
    load_b2(pw, c0, c1, half * 16, b);
    __syncthreads();

    for (int ks = 0; ks < 64; ++ks) {
        const int buf = ks & 1;
        Frag a[4];
#pragma unroll
        for (int m = 0; m < 4; ++m) {
            const unsigned short* ap = As[buf] + (m * 16 + lrow) * 32 + half * 16;
            a[m].u[0] = *(const uint4*)ap;
            a[m].u[1] = *(const uint4*)(ap + 8);
        }
        uint4 stgN;
        Frag bn[2];
        if (ks < 63) {
            stgN = *(const uint4*)(stgp + (ks + 1) * 32);
            load_b2(pw, c0, c1, (ks + 1) * 32 + half * 16, bn);
        }
#pragma unroll
        for (int m = 0; m < 4; ++m) {
            acc[m][0] = wmma_bf16(a[m], b[0], acc[m][0]);
            acc[m][1] = wmma_bf16(a[m], b[1], acc[m][1]);
        }
        if (ks < 63) {
            *(uint4*)(As[buf ^ 1] + srow * 32 + schunk) = stgN;
            __syncthreads();
            b[0] = bn[0]; b[1] = bn[1];
        }
    }

    const float pb0 = pb[c0], pb1 = pb[c1];
#pragma unroll
    for (int m = 0; m < 4; ++m)
#pragma unroll
        for (int v = 0; v < 8; ++v) {
            const size_t r = (size_t)(mbase + m * 16 + v + half * 8);
            out[r * FDIM + c0] = acc[m][0][v] + pb0;
            out[r * FDIM + c1] = acc[m][1][v] + pb1;
        }
}

// ---------------- host ----------------

extern "C" void kernel_launch(void* const* d_in, const int* in_sizes, int n_in,
                              void* d_out, int out_size, void* d_ws, size_t ws_size,
                              hipStream_t stream) {
    const int*   char_ids   = (const int*)d_in[0];
    const float* char_table = (const float*)d_in[1];
    const float* cw[7]; const float* cb[7];
    for (int i = 0; i < 7; ++i) { cw[i] = (const float*)d_in[2 + 2 * i]; cb[i] = (const float*)d_in[3 + 2 * i]; }
    const float* hw0_tw = (const float*)d_in[16]; const float* hw0_tb = (const float*)d_in[17];
    const float* hw0_gw = (const float*)d_in[18]; const float* hw0_gb = (const float*)d_in[19];
    const float* hw1_tw = (const float*)d_in[20]; const float* hw1_tb = (const float*)d_in[21];
    const float* hw1_gw = (const float*)d_in[22]; const float* hw1_gb = (const float*)d_in[23];
    const float* proj_w = (const float*)d_in[24]; const float* proj_b = (const float*)d_in[25];

    uint8_t* ws = (uint8_t*)d_ws;
    size_t off = 0;
    auto alloc = [&](size_t bytes) -> void* {
        off = (off + 255) & ~(size_t)255;
        void* p = ws + off; off += bytes; return p;
    };
    unsigned short* wcat = (unsigned short*)alloc((size_t)FDIM * KC * 2);
    float*          bcat = (float*)alloc((size_t)FDIM * 4);
    unsigned short* twb0 = (unsigned short*)alloc((size_t)FDIM * FDIM * 2);
    unsigned short* gwb0 = (unsigned short*)alloc((size_t)FDIM * FDIM * 2);
    unsigned short* twb1 = (unsigned short*)alloc((size_t)FDIM * FDIM * 2);
    unsigned short* gwb1 = (unsigned short*)alloc((size_t)FDIM * FDIM * 2);
    unsigned short* pwb  = (unsigned short*)alloc((size_t)FDIM * FDIM * 2);
    float*          hfA  = (float*)alloc((size_t)NTOK * FDIM * 4);
    unsigned short* hbA  = (unsigned short*)alloc((size_t)NTOK * FDIM * 2);
    float*          hfB  = (float*)alloc((size_t)NTOK * FDIM * 4);
    unsigned short* hbB  = (unsigned short*)alloc((size_t)NTOK * FDIM * 2);
    (void)ws_size; (void)in_sizes; (void)n_in; (void)out_size;

    const int nW = FDIM * FDIM;
    cvt_bf16_kernel<<<1024, 256, 0, stream>>>(hw0_tw, twb0, nW);
    cvt_bf16_kernel<<<1024, 256, 0, stream>>>(hw0_gw, gwb0, nW);
    cvt_bf16_kernel<<<1024, 256, 0, stream>>>(hw1_tw, twb1, nW);
    cvt_bf16_kernel<<<1024, 256, 0, stream>>>(hw1_gw, gwb1, nW);
    cvt_bf16_kernel<<<1024, 256, 0, stream>>>(proj_w, pwb, nW);
    build_wcat_kernel<<<(FDIM * KC + 255) / 256, 256, 0, stream>>>(
        cw[0], cw[1], cw[2], cw[3], cw[4], cw[5], cw[6],
        cb[0], cb[1], cb[2], cb[3], cb[4], cb[5], cb[6], wcat, bcat);

    embed_conv_kernel<<<NTOK, 256, 0, stream>>>(char_ids, char_table, wcat, bcat, hfA, hbA);

    dim3 ggrid(FDIM / 256, NTOK / 64);
    highway_kernel<<<ggrid, 256, 0, stream>>>(hbA, hfA, twb0, hw0_tb, gwb0, hw0_gb, hfB, hbB);
    highway_kernel<<<ggrid, 256, 0, stream>>>(hbB, hfB, twb1, hw1_tb, gwb1, hw1_gb, hfA, hbA);
    proj_kernel<<<ggrid, 256, 0, stream>>>(hbA, pwb, proj_b, (float*)d_out);
}
